// SymmetrizationNetwork_19035295055965
// MI455X (gfx1250) — matrix operand order repaired and checked by reading, not compiled, not run on hardware
//
#include <hip/hip_runtime.h>

#ifndef __has_builtin
#define __has_builtin(x) 0
#endif

typedef __attribute__((ext_vector_type(16))) _Float16 v16h;
typedef __attribute__((ext_vector_type(8)))  _Float16 v8h;
typedef __attribute__((ext_vector_type(8)))  float    v8f;
typedef __attribute__((ext_vector_type(4)))  float    v4f;
typedef __attribute__((ext_vector_type(4)))  unsigned int v4u;

// exact pointee type expected by the async-LDS builtin (from hipcc diagnostic)
typedef int b128i __attribute__((vector_size(16)));
typedef __attribute__((address_space(1))) b128i* gas_b128_t;
typedef __attribute__((address_space(3))) b128i* las_b128_t;

#define P_ 16
#define B_ 16
#define N_ 32
#define L_ 16384
#define H_ 64
#define TPW 2                 /* tiles (of 16 positions) per wave   */
#define WAVES 8
#define POS_PER_WG (WAVES*TPW*16)   /* 256 positions per workgroup  */
#define PREP_BLOCKS 32

#if defined(__AMDGCN__) && __has_builtin(__builtin_amdgcn_global_load_async_to_lds_b128)
#define USE_ASYNC_LDS 1
#else
#define USE_ASYNC_LDS 0
#endif

// ---------------------------------------------------------------------------
// Prep: fold perms into weights, emit f16 WMMA A-fragments in per-lane layout.
// A-fragment (16-bit A 16x32, wave32): lane L holds row M=L&15; half j maps to
// K = j + 16*(L>>4) (contiguous k order, same as the B fragments).
// Parallelized over PREP_BLOCKS blocks; each block rebuilds the tiny inv table.
// ---------------------------------------------------------------------------
__global__ __launch_bounds__(256) void prep_kernel(
    const long long* __restrict__ perms, const float* __restrict__ W1,
    const float* __restrict__ W2, const float* __restrict__ b2,
    _Float16* __restrict__ A1, _Float16* __restrict__ A2,
    float* __restrict__ bconst)
{
  __shared__ int inv[P_ * N_];
  const int tid = threadIdx.x;
  for (int idx = tid; idx < P_ * N_; idx += 256) {
    int p = idx >> 5, n = idx & 31;
    int m = (int)perms[p * N_ + n];
    inv[p * N_ + m] = n;
  }
  __syncthreads();
  const int gstart = blockIdx.x * 256 + tid;
  const int gstep  = PREP_BLOCKS * 256;
  // Layer-1 A fragments: [p][t(0..3)][lane][j] ; A1[hidden,K] = W1[hidden, inv_p[K]]
  for (int idx = gstart; idx < P_ * 4 * 32 * 16; idx += gstep) {
    int p    = idx >> 11;
    int t    = (idx >> 9) & 3;
    int lane = (idx >> 4) & 31;
    int j    = idx & 15;
    int hidden = 16 * t + (lane & 15);
    int K      = j + 16 * (lane >> 4);
    A1[idx] = (_Float16)W1[hidden * N_ + inv[p * N_ + K]];
  }
  // Layer-2 A fragments: [p][u(0..1)][c(0..1)][lane][j] ;
  // A2[n, 32c+K] = W2[inv_p[n], 32c+K] / P
  for (int idx = gstart; idx < P_ * 2 * 2 * 32 * 16; idx += gstep) {
    int p    = idx >> 11;
    int u    = (idx >> 10) & 1;
    int c    = (idx >> 9) & 1;
    int lane = (idx >> 4) & 31;
    int j    = idx & 15;
    int n      = 16 * u + (lane & 15);
    int K      = j + 16 * (lane >> 4);
    int hidden = 32 * c + K;
    A2[idx] = (_Float16)(W2[inv[p * N_ + n] * H_ + hidden] * (1.0f / (float)P_));
  }
  if (blockIdx.x == 0 && tid < N_) {
    float s = 0.f;
    for (int p = 0; p < P_; ++p) s += b2[inv[p * N_ + tid]];
    bconst[tid] = s * (1.0f / (float)P_);
  }
}

// ---------------------------------------------------------------------------
// Main kernel: per wave, out^T(32x16) = sum_p W2fold_p x relu(W1fold_p x x^T + b1)
// ---------------------------------------------------------------------------
__global__ __launch_bounds__(256, 2) void mlp_kernel(
    const float* __restrict__ x, const float* __restrict__ b1,
    const _Float16* __restrict__ A1g, const _Float16* __restrict__ A2g,
    const float* __restrict__ bconst, float* __restrict__ out)
{
  __shared__ __align__(32) _Float16 wAll[65536];          // 128 KB folded weights
  __shared__ __align__(32) _Float16 xT[POS_PER_WG * N_];  // 16 KB  x^T as f16 [pos][ch]
  __shared__ __align__(32) float    scratch[WAVES * 512]; // 2 KB per wave

  const int tid   = threadIdx.x;
  const int b     = blockIdx.y;
  const int Lbase = blockIdx.x * POS_PER_WG;

  // ---- stage folded weights (A1g..A2g are contiguous: 131072 bytes) ----
#if USE_ASYNC_LDS
  {
    const char* gsrc = (const char*)A1g;
    char*       ldst = (char*)wAll;
    #pragma unroll
    for (int k = 0; k < 32; ++k) {
      int off = (tid + k * 256) * 16;
      __builtin_amdgcn_global_load_async_to_lds_b128(
          (gas_b128_t)(gsrc + off),
          (las_b128_t)(ldst + off),
          0, 0);
    }
#if __has_builtin(__builtin_amdgcn_s_wait_asynccnt)
    __builtin_amdgcn_s_wait_asynccnt(0);
#else
    asm volatile("s_wait_asynccnt 0" ::: "memory");
#endif
  }
#else
  {
    const v4u* s1 = (const v4u*)A1g;
    v4u*       d1 = (v4u*)wAll;
    #pragma unroll
    for (int k = 0; k < 32; ++k) {
      int i = tid + k * 256;
      d1[i] = s1[i];
    }
  }
#endif

  // ---- stage x chunk transposed to f16: xT[pos][ch] ----
  #pragma unroll
  for (int k = 0; k < 8; ++k) {
    int idx = tid + k * 256;               // 2048 float4s = 32 ch x 64 float4
    int c   = idx >> 6;
    int q   = idx & 63;
    v4f v = *(const v4f*)&x[((size_t)b * N_ + c) * L_ + Lbase + 4 * q];
    int pos = 4 * q;
    xT[(pos + 0) * N_ + c] = (_Float16)v.x;
    xT[(pos + 1) * N_ + c] = (_Float16)v.y;
    xT[(pos + 2) * N_ + c] = (_Float16)v.z;
    xT[(pos + 3) * N_ + c] = (_Float16)v.w;
  }
  __syncthreads();

  const int wave = tid >> 5;
  const int lane = tid & 31;
  const int lhi  = lane >> 4;
  const int llo  = lane & 15;

  // biases in C/D fragment layout: value (reg r, lane) -> row M = r + 8*lhi
  v8f bC1[4];
  #pragma unroll
  for (int tt = 0; tt < 4; ++tt) {
    const float* src = &b1[16 * tt + 8 * lhi];
    #pragma unroll
    for (int r = 0; r < 8; ++r) bC1[tt][r] = src[r];
  }
  v8f bOut[2];
  #pragma unroll
  for (int u = 0; u < 2; ++u) {
    const float* src = &bconst[16 * u + 8 * lhi];
    #pragma unroll
    for (int r = 0; r < 8; ++r) bOut[u][r] = src[r];
  }

  _Float16* hbuf = (_Float16*)&scratch[wave * 512];  // per-wave [16 pos][64 hidden] f16

  v8f acc[TPW][2];
  #pragma unroll
  for (int t = 0; t < TPW; ++t)
    #pragma unroll
    for (int u = 0; u < 2; ++u)
      #pragma unroll
      for (int r = 0; r < 8; ++r) acc[t][u][r] = 0.f;

  const _Float16* wA1 = wAll;
  const _Float16* wA2 = wAll + 32768;

  for (int p = 0; p < P_; ++p) {
    v16h a1[4], a2[2][2];
    #pragma unroll
    for (int tt = 0; tt < 4; ++tt)
      a1[tt] = *(const v16h*)&wA1[((p * 4 + tt) * 32 + lane) * 16];
    #pragma unroll
    for (int u = 0; u < 2; ++u)
      #pragma unroll
      for (int c = 0; c < 2; ++c)
        a2[u][c] = *(const v16h*)&wA2[(((p * 2 + u) * 2 + c) * 32 + lane) * 16];

    #pragma unroll
    for (int t = 0; t < TPW; ++t) {
      const int tb = (wave * TPW + t) * 16;
      // B = x^T (32x16): lane holds pos=llo, K = 16*lhi + j -> contiguous in xT
      v16h bfrag = *(const v16h*)&xT[(tb + llo) * N_ + 16 * lhi];

      // layer 1: h^T tiles (hidden 16t..16t+15) x (16 positions), K = 32 channels
      v8f c1[4];
      #pragma unroll
      for (int tt = 0; tt < 4; ++tt)
        c1[tt] = __builtin_amdgcn_wmma_f32_16x16x32_f16(
            false, a1[tt], false, bfrag, (short)0, bC1[tt], false, false);

      // cvt to f16 first (canonical result), then relu as packed f16 max:
      // max(cvt(x),0) == cvt(max(x,0)) exactly; avoids canonicalize+max on f32
      #pragma unroll
      for (int tt = 0; tt < 4; ++tt) {
        v8h h8;
        #pragma unroll
        for (int r = 0; r < 8; ++r)
          h8[r] = (_Float16)c1[tt][r];
        v8h zero8 = {};
        h8 = __builtin_elementwise_max(h8, zero8);
        // value (tt, r, lane): hidden = 16tt + 8*lhi + r, pos = llo
        *(v8h*)&hbuf[llo * 64 + 16 * tt + 8 * lhi] = h8;
      }

      // layer 2: out^T tiles (n 16u..16u+15) x (16 positions), K = 64 hidden
      #pragma unroll
      for (int c = 0; c < 2; ++c) {
        v16h b2f = *(const v16h*)&hbuf[llo * 64 + 32 * c + 16 * lhi];
        #pragma unroll
        for (int u = 0; u < 2; ++u)
          acc[t][u] = __builtin_amdgcn_wmma_f32_16x16x32_f16(
              false, a2[u][c], false, b2f, (short)0, acc[t][u], false, false);
      }
    }
  }

  // ---- epilogue: add output bias, transpose via LDS, coalesced global store ----
  #pragma unroll
  for (int t = 0; t < TPW; ++t) {
    const int tb = (wave * TPW + t) * 16;
    float* so = &scratch[wave * 512];  // reuse as [n(32)][pos(16)] f32
    #pragma unroll
    for (int u = 0; u < 2; ++u)
      #pragma unroll
      for (int r = 0; r < 8; ++r) {
        int n = 16 * u + 8 * lhi + r;
        so[n * 16 + llo] = acc[t][u][r] + bOut[u][r];
      }
    // lane = n row; 16 contiguous f32 per row -> 4x b128 stores
    const float* rowsrc = &so[lane * 16];
    float* gdst = &out[((size_t)b * N_ + lane) * L_ + Lbase + tb];
    #pragma unroll
    for (int k = 0; k < 4; ++k)
      *(v4f*)&gdst[4 * k] = *(const v4f*)&rowsrc[4 * k];
  }
}

// ---------------------------------------------------------------------------
extern "C" void kernel_launch(void* const* d_in, const int* in_sizes, int n_in,
                              void* d_out, int out_size, void* d_ws, size_t ws_size,
                              hipStream_t stream) {
  (void)in_sizes; (void)n_in; (void)out_size; (void)ws_size;
  const float*     x     = (const float*)d_in[0];
  const long long* perms = (const long long*)d_in[1];
  const float*     W1    = (const float*)d_in[2];
  const float*     b1    = (const float*)d_in[3];
  const float*     W2    = (const float*)d_in[4];
  const float*     b2    = (const float*)d_in[5];
  float* out = (float*)d_out;

  _Float16* A1 = (_Float16*)d_ws;          // 65536 B
  _Float16* A2 = A1 + 32768;               // 65536 B (contiguous with A1)
  float* bconst = (float*)((char*)d_ws + 131072);  // 128 B

  prep_kernel<<<PREP_BLOCKS, 256, 0, stream>>>(perms, W1, W2, b2, A1, A2, bconst);

  dim3 grid(L_ / POS_PER_WG, B_);
  mlp_kernel<<<grid, 256, 0, stream>>>(x, b1, A1, A2, bconst, out);
}
